// MambaLayer_43525198578344
// MI455X (gfx1250) — matrix operand order repaired and checked by reading, not compiled, not run on hardware
//
#include <hip/hip_runtime.h>

// ---------------------------------------------------------------------------
// Mamba layer for MI455X (gfx1250, wave32).
// Pipeline: transpose weights->bf16 | rmsnorm->bf16 | WMMA GEMM1 (w_in)
//           | causal depthwise conv + silu | WMMA GEMM2 (w_x) | WMMA GEMM3
//           (w_dt, +bias+softplus) | chunked selective scan (3 passes)
//           + gating fused | WMMA GEMM4 (w_out, +residual) -> d_out.
// All matmuls use v_wmma_f32_16x16x32_bf16 with f32 accumulation.
// ---------------------------------------------------------------------------

#define D_MODEL  1024
#define D_INNER  2048
#define D_STATE  16
#define DT_RANK  64
#define BB       2
#define TT       4096
#define ROWS     (BB * TT)          // 8192
#define NCHUNK   32
#define CHUNK    128                // TT / NCHUNK

typedef __attribute__((ext_vector_type(16))) __bf16 bf16x16;
typedef __attribute__((ext_vector_type(8)))  float  f32x8;

union FragBF { bf16x16 v; uint4 q[2]; };

__device__ __forceinline__ unsigned short f2bf(float f) {
  unsigned u = __float_as_uint(f);
  u += 0x7FFFu + ((u >> 16) & 1u);          // round-to-nearest-even
  return (unsigned short)(u >> 16);
}
__device__ __forceinline__ float softplus_f(float x) {
  return x > 20.f ? x : log1pf(__expf(x));
}
__device__ __forceinline__ float silu_f(float x) {
  return x / (1.f + __expf(-x));
}

// ---------------------------------------------------------------------------
// Tiled transpose fp32 (K,N) -> bf16 (Npad,K); pad rows [N,Npad) with zeros.
// ---------------------------------------------------------------------------
__global__ __launch_bounds__(256)
void transpose_bf16(const float* __restrict__ in, unsigned short* __restrict__ out,
                    int K, int N, int Npad) {
  __shared__ float tile[32][33];
  const int k0 = blockIdx.x * 32, n0 = blockIdx.y * 32;
  const int tx = threadIdx.x, ty = threadIdx.y;
#pragma unroll
  for (int j = 0; j < 4; j++) {
    int k = k0 + ty + j * 8;
    int n = n0 + tx;
    tile[ty + j * 8][tx] = (n < N) ? in[(size_t)k * N + n] : 0.f;
  }
  __syncthreads();
#pragma unroll
  for (int j = 0; j < 4; j++) {
    int n = n0 + ty + j * 8;
    int k = k0 + tx;
    if (n < Npad) out[(size_t)n * K + k] = f2bf(tile[tx][ty + j * 8]);
  }
}

__global__ __launch_bounds__(256)
void aneg_k(const float* __restrict__ A_log, float* __restrict__ Aneg) {
  int id = blockIdx.x * 256 + threadIdx.x;
  if (id < D_INNER * D_STATE) Aneg[id] = -__expf(A_log[id]);
}

// ---------------------------------------------------------------------------
// RMSNorm (one block per token row) -> bf16 activations.
// ---------------------------------------------------------------------------
__global__ __launch_bounds__(256)
void rmsnorm_k(const float* __restrict__ x, const float* __restrict__ w,
               unsigned short* __restrict__ xn) {
  const int row = blockIdx.x, tid = threadIdx.x;
  const float* xr = x + (size_t)row * D_MODEL;
  float4 v = *(const float4*)(xr + tid * 4);
  float p = v.x * v.x + v.y * v.y + v.z * v.z + v.w * v.w;
  __shared__ float red[256];
  red[tid] = p;
  __syncthreads();
  for (int s = 128; s > 0; s >>= 1) {
    if (tid < s) red[tid] += red[tid + s];
    __syncthreads();
  }
  const float scale = rsqrtf(red[0] * (1.f / D_MODEL) + 1e-5f);
  const float4 wv = *(const float4*)(w + tid * 4);
  unsigned short* o = xn + (size_t)row * D_MODEL + tid * 4;
  o[0] = f2bf(v.x * scale * wv.x);
  o[1] = f2bf(v.y * scale * wv.y);
  o[2] = f2bf(v.z * scale * wv.z);
  o[3] = f2bf(v.w * scale * wv.w);
}

// ---------------------------------------------------------------------------
// bf16 WMMA GEMM: C(M,N) = A(M,K) * Bt(N,K)^T.  Block tile 128x128, 8 waves
// (4 along M x 2 along N), wave tile 32x64 = 2x4 wmma accumulators.
// EPI: 0 plain | 1 also bf16-pack cols<64 (dt) | 2 +bias, softplus | 3 +resid
// ---------------------------------------------------------------------------
template <int EPI>
__global__ __launch_bounds__(256)
void gemm_bf16_k(const unsigned short* __restrict__ A,
                 const unsigned short* __restrict__ Bt,
                 float* __restrict__ C, const int M, const int N, const int K,
                 const float* __restrict__ bias, const float* __restrict__ resid,
                 unsigned short* __restrict__ obf) {
  constexpr int PITCH = 40;  // halves; 80B row pitch -> conflict-free b128
  __shared__ __align__(16) unsigned short As[128 * PITCH];
  __shared__ __align__(16) unsigned short Bs[128 * PITCH];

  const int tid = threadIdx.x;
  const int lane = tid & 31, wave = tid >> 5;
  const int wm = (wave & 3) * 32;   // wave M offset in block
  const int wn = (wave >> 2) * 64;  // wave N offset in block
  const int bm = blockIdx.x * 128, bn = blockIdx.y * 128;

  const int lrow = tid >> 1;         // 0..127 (tile row)
  const int lseg = (tid & 1) * 16;   // half-row segment (halves)
  const unsigned short* gA = A + (size_t)(bm + lrow) * K + lseg;
  const unsigned short* gB = Bt + (size_t)(bn + lrow) * K + lseg;
  unsigned short* sA = &As[lrow * PITCH + lseg];
  unsigned short* sB = &Bs[lrow * PITCH + lseg];

  const int l16 = lane & 15, lg = lane >> 4;

  f32x8 acc[2][4];
  const f32x8 zero = {0.f, 0.f, 0.f, 0.f, 0.f, 0.f, 0.f, 0.f};
#pragma unroll
  for (int i = 0; i < 2; i++)
#pragma unroll
    for (int j = 0; j < 4; j++) acc[i][j] = zero;

  for (int k0 = 0; k0 < K; k0 += 32) {
    uint4 a0 = *(const uint4*)(gA + k0);
    uint4 a1 = *(const uint4*)(gA + k0 + 8);
    uint4 b0 = *(const uint4*)(gB + k0);
    uint4 b1 = *(const uint4*)(gB + k0 + 8);
    __syncthreads();
    *(uint4*)(sA) = a0; *(uint4*)(sA + 8) = a1;
    *(uint4*)(sB) = b0; *(uint4*)(sB + 8) = b1;
    __syncthreads();

    FragBF af[2], bf[4];
#pragma unroll
    for (int mt = 0; mt < 2; mt++) {
      // A 16x32 bf16: lane l -> row l%16, K halves {8g..}+{16+8g..}
      const unsigned short* p = &As[(wm + mt * 16 + l16) * PITCH + lg * 8];
      af[mt].q[0] = *(const uint4*)(p);
      af[mt].q[1] = *(const uint4*)(p + 16);
    }
#pragma unroll
    for (int nt = 0; nt < 4; nt++) {
      // B 32x16 bf16: lane l -> col l%16, K halves {8g..}+{16+8g..}
      const unsigned short* p = &Bs[(wn + nt * 16 + l16) * PITCH + lg * 8];
      bf[nt].q[0] = *(const uint4*)(p);
      bf[nt].q[1] = *(const uint4*)(p + 16);
    }
#pragma unroll
    for (int mt = 0; mt < 2; mt++)
#pragma unroll
      for (int nt = 0; nt < 4; nt++)
        acc[mt][nt] = __builtin_amdgcn_wmma_f32_16x16x32_bf16(
            false, af[mt].v, false, bf[nt].v, (short)0, acc[mt][nt], false, false);
  }

  // Epilogue.  C/D layout: vgpr i, lane l -> row i+8*(l/16), col l%16.
#pragma unroll
  for (int mt = 0; mt < 2; mt++) {
#pragma unroll
    for (int nt = 0; nt < 4; nt++) {
#pragma unroll
      for (int i = 0; i < 8; i++) {
        int row = bm + wm + mt * 16 + i + 8 * lg;
        int col = bn + wn + nt * 16 + l16;
        if (col < N) {
          float v = acc[mt][nt][i];
          if (EPI == 2) v = softplus_f(v + bias[col]);
          if (EPI == 3) v += resid[(size_t)row * N + col];
          C[(size_t)row * N + col] = v;
          if (EPI == 1 && col < DT_RANK)
            obf[(size_t)row * DT_RANK + col] = f2bf(v);
        }
      }
    }
  }
}

// ---------------------------------------------------------------------------
// Causal depthwise conv (K=4) + bias + SiLU.  u lives in cols [0,2048) of xr.
// ---------------------------------------------------------------------------
__global__ __launch_bounds__(256)
void conv_silu_k(const float* __restrict__ xr, const float* __restrict__ cw,
                 const float* __restrict__ cb, float* __restrict__ uf,
                 unsigned short* __restrict__ ubf) {
  const size_t id = (size_t)blockIdx.x * 256 + threadIdx.x;  // B*T*D_INNER
  const int d = (int)(id & (D_INNER - 1));
  const int t = (int)((id >> 11) & (TT - 1));
  const int b = (int)(id >> 23);
  float acc = cb[d];
#pragma unroll
  for (int j = 0; j < 4; j++) {
    int tt = t - 3 + j;
    if (tt >= 0)
      acc += cw[d * 4 + j] * xr[((size_t)(b * TT + tt)) * (2 * D_INNER) + d];
  }
  const float s = silu_f(acc);
  uf[id] = s;
  ubf[id] = f2bf(s);
}

// ---------------------------------------------------------------------------
// Selective scan, chunked (associative: h_t = a_t*h + b_t).
// Pass 1: per-chunk local scan from h=0 -> store h_end and prod(a).
// ---------------------------------------------------------------------------
__global__ __launch_bounds__(256)
void scan1_k(const float* __restrict__ delta, const float* __restrict__ u,
             const float* __restrict__ xdbc, const float* __restrict__ Aneg,
             float* __restrict__ hend, float* __restrict__ aprod) {
  const int b = blockIdx.x >> 8;           // 256 blocks per batch
  const int rem = blockIdx.x & 255;
  const int c = rem >> 3;                  // chunk
  const int d = ((rem & 7) << 8) + threadIdx.x;
  __shared__ float bs[CHUNK * D_STATE];    // Bm for this chunk
  for (int i = threadIdx.x; i < CHUNK * D_STATE; i += 256) {
    int t = i >> 4, n = i & 15;
    bs[i] = xdbc[((size_t)(b * TT + c * CHUNK + t)) * 96 + 64 + n];
  }
  __syncthreads();
  float Al[D_STATE];
#pragma unroll
  for (int n = 0; n < D_STATE; n++)
    Al[n] = Aneg[d * D_STATE + n] * 1.4426950408889634f;  // * log2(e)
  float h[D_STATE], ap[D_STATE];
#pragma unroll
  for (int n = 0; n < D_STATE; n++) { h[n] = 0.f; ap[n] = 1.f; }
  for (int t = 0; t < CHUNK; t++) {
    const size_t idx = ((size_t)(b * TT + c * CHUNK + t)) * D_INNER + d;
    const float dl = delta[idx];
    const float du = dl * u[idx];
#pragma unroll
    for (int n = 0; n < D_STATE; n++) {
      float e = exp2f(dl * Al[n]);
      h[n] = e * h[n] + du * bs[(t << 4) + n];
      ap[n] *= e;
    }
  }
#pragma unroll
  for (int n = 0; n < D_STATE; n++) {
    size_t o = (((size_t)(b * NCHUNK + c) * D_STATE) + n) * D_INNER + d;
    hend[o] = h[n];
    aprod[o] = ap[n];
  }
}

// Pass 2: tiny sequential combine across the 32 chunks; rewrites hend with
// each chunk's *initial* state.
__global__ __launch_bounds__(256)
void scan2_k(float* __restrict__ hend, const float* __restrict__ aprod) {
  const int id = blockIdx.x * 256 + threadIdx.x;  // B*D_INNER = 4096
  const int b = id >> 11, d = id & (D_INNER - 1);
  float h[D_STATE];
#pragma unroll
  for (int n = 0; n < D_STATE; n++) h[n] = 0.f;
  for (int c = 0; c < NCHUNK; c++) {
#pragma unroll
    for (int n = 0; n < D_STATE; n++) {
      size_t o = (((size_t)(b * NCHUNK + c) * D_STATE) + n) * D_INNER + d;
      float e = hend[o];
      float a = aprod[o];
      hend[o] = h[n];            // becomes h_init for chunk c
      h[n] = a * h[n] + e;
    }
  }
}

// Pass 3: rescan each chunk from h_init, emit y, fuse +u*D and *silu(res),
// pack bf16 for the output GEMM.
__global__ __launch_bounds__(256)
void scan3_k(const float* __restrict__ delta, const float* __restrict__ u,
             const float* __restrict__ xdbc, const float* __restrict__ Aneg,
             const float* __restrict__ hinit, const float* __restrict__ xr,
             const float* __restrict__ Dp, unsigned short* __restrict__ ybf) {
  const int b = blockIdx.x >> 8;
  const int rem = blockIdx.x & 255;
  const int c = rem >> 3;
  const int d = ((rem & 7) << 8) + threadIdx.x;
  __shared__ float bs[CHUNK * 32];  // Bm | Cm for this chunk
  for (int i = threadIdx.x; i < CHUNK * 32; i += 256) {
    int t = i >> 5, n = i & 31;
    bs[i] = xdbc[((size_t)(b * TT + c * CHUNK + t)) * 96 + 64 + n];
  }
  __syncthreads();
  float Al[D_STATE], h[D_STATE];
#pragma unroll
  for (int n = 0; n < D_STATE; n++) {
    Al[n] = Aneg[d * D_STATE + n] * 1.4426950408889634f;
    h[n] = hinit[(((size_t)(b * NCHUNK + c) * D_STATE) + n) * D_INNER + d];
  }
  const float dpd = Dp[d];
  for (int t = 0; t < CHUNK; t++) {
    const size_t idx = ((size_t)(b * TT + c * CHUNK + t)) * D_INNER + d;
    const float dl = delta[idx];
    const float uu = u[idx];
    const float du = dl * uu;
    float y = 0.f;
#pragma unroll
    for (int n = 0; n < D_STATE; n++) {
      float e = exp2f(dl * Al[n]);
      h[n] = e * h[n] + du * bs[(t << 5) + n];
      y += h[n] * bs[(t << 5) + 16 + n];
    }
    const float res =
        xr[((size_t)(b * TT + c * CHUNK + t)) * (2 * D_INNER) + D_INNER + d];
    ybf[idx] = f2bf((y + uu * dpd) * silu_f(res));
  }
}

// ---------------------------------------------------------------------------
extern "C" void kernel_launch(void* const* d_in, const int* in_sizes, int n_in,
                              void* d_out, int out_size, void* d_ws,
                              size_t ws_size, hipStream_t stream) {
  (void)in_sizes; (void)n_in; (void)out_size; (void)ws_size;
  const float* x      = (const float*)d_in[0];
  const float* norm_w = (const float*)d_in[1];
  const float* w_in   = (const float*)d_in[2];
  const float* conv_w = (const float*)d_in[3];
  const float* conv_b = (const float*)d_in[4];
  const float* w_x    = (const float*)d_in[5];
  const float* w_dt   = (const float*)d_in[6];
  const float* b_dt   = (const float*)d_in[7];
  const float* A_log  = (const float*)d_in[8];
  const float* Dp     = (const float*)d_in[9];
  const float* w_out  = (const float*)d_in[10];

  char* ws = (char*)d_ws;
  size_t off = 0;
  auto alloc = [&](size_t bytes) {
    size_t o = off;
    off = (off + bytes + 255) & ~(size_t)255;
    return o;
  };
  unsigned short* w_inT  = (unsigned short*)(ws + alloc((size_t)4096 * 1024 * 2));
  unsigned short* w_xT   = (unsigned short*)(ws + alloc((size_t)128 * 2048 * 2));
  unsigned short* w_dtT  = (unsigned short*)(ws + alloc((size_t)2048 * 64 * 2));
  unsigned short* w_outT = (unsigned short*)(ws + alloc((size_t)1024 * 2048 * 2));
  float* Aneg            = (float*)(ws + alloc((size_t)D_INNER * D_STATE * 4));
  unsigned short* xn_bf  = (unsigned short*)(ws + alloc((size_t)ROWS * 1024 * 2));
  float* xr              = (float*)(ws + alloc((size_t)ROWS * 4096 * 4));
  float* u_f             = (float*)(ws + alloc((size_t)ROWS * 2048 * 4));
  unsigned short* u_bf   = (unsigned short*)(ws + alloc((size_t)ROWS * 2048 * 2));
  float* xdbc            = (float*)(ws + alloc((size_t)ROWS * 96 * 4));
  unsigned short* dt_bf  = (unsigned short*)(ws + alloc((size_t)ROWS * 64 * 2));
  float* delta           = (float*)(ws + alloc((size_t)ROWS * 2048 * 4));
  float* hend            = (float*)(ws + alloc((size_t)BB * NCHUNK * 16 * 2048 * 4));
  float* aprod           = (float*)(ws + alloc((size_t)BB * NCHUNK * 16 * 2048 * 4));
  unsigned short* y_bf   = (unsigned short*)(ws + alloc((size_t)ROWS * 2048 * 2));

  const dim3 tb(32, 8, 1);
  transpose_bf16<<<dim3(1024 / 32, 4096 / 32), tb, 0, stream>>>(w_in, w_inT, 1024, 4096, 4096);
  transpose_bf16<<<dim3(2048 / 32, 128 / 32), tb, 0, stream>>>(w_x, w_xT, 2048, 96, 128);
  transpose_bf16<<<dim3(64 / 32, 2048 / 32), tb, 0, stream>>>(w_dt, w_dtT, 64, 2048, 2048);
  transpose_bf16<<<dim3(2048 / 32, 1024 / 32), tb, 0, stream>>>(w_out, w_outT, 2048, 1024, 1024);
  aneg_k<<<(D_INNER * D_STATE) / 256, 256, 0, stream>>>(A_log, Aneg);

  rmsnorm_k<<<ROWS, 256, 0, stream>>>(x, norm_w, xn_bf);

  gemm_bf16_k<0><<<dim3(ROWS / 128, 4096 / 128), 256, 0, stream>>>(
      xn_bf, w_inT, xr, ROWS, 4096, 1024, nullptr, nullptr, nullptr);

  conv_silu_k<<<(ROWS * D_INNER) / 256, 256, 0, stream>>>(xr, conv_w, conv_b, u_f, u_bf);

  gemm_bf16_k<1><<<dim3(ROWS / 128, 1), 256, 0, stream>>>(
      u_bf, w_xT, xdbc, ROWS, 96, 2048, nullptr, nullptr, dt_bf);

  gemm_bf16_k<2><<<dim3(ROWS / 128, 2048 / 128), 256, 0, stream>>>(
      dt_bf, w_dtT, delta, ROWS, 2048, 64, b_dt, nullptr, nullptr);

  scan1_k<<<BB * 256, 256, 0, stream>>>(delta, u_f, xdbc, Aneg, hend, aprod);
  scan2_k<<<(BB * D_INNER) / 256, 256, 0, stream>>>(hend, aprod);
  scan3_k<<<BB * 256, 256, 0, stream>>>(delta, u_f, xdbc, Aneg, hend, xr, Dp, y_bf);

  gemm_bf16_k<3><<<dim3(ROWS / 128, 1024 / 128), 256, 0, stream>>>(
      y_bf, w_outT, (float*)d_out, ROWS, 1024, 2048, nullptr, x, nullptr);
}
